// BitNetDecoder_31224412242445
// MI455X (gfx1250) — 8-bit matrix operand order repaired and checked by reading, compiled, not run on hardware
//
#include <hip/hip_runtime.h>
#include <math.h>
#include <stdint.h>
#include <stddef.h>

#define DMODEL 1024
#define NHEAD  16
#define HDIM   64
#define FDIM   4096
#define NLAYER 8
#define BATCH  2
#define SEQ    1024
#define NTOK   (BATCH*SEQ)
#define VOCAB  32000

typedef signed char     i8;
typedef unsigned short  u16;
typedef __attribute__((ext_vector_type(8)))  int      v8i;
typedef __attribute__((ext_vector_type(8)))  float    v8f;
typedef __attribute__((ext_vector_type(16))) _Float16 v16h;
typedef __attribute__((ext_vector_type(16))) __bf16   v16bf;

union FragI { v8i  v; int i[8]; int2 d[4]; int4 q[2]; };
union FragH { v16h v; int i[8]; int4 q[2]; };
union FragB { v16bf v; int i[8]; int4 q[2]; };

__global__ void k_embed(const int* __restrict__ ids, const float* __restrict__ emb,
                        const float* __restrict__ pos, float* __restrict__ x) {
  size_t total = (size_t)NTOK * DMODEL;
  for (size_t idx = (size_t)blockIdx.x * blockDim.x + threadIdx.x; idx < total;
       idx += (size_t)gridDim.x * blockDim.x) {
    int tok = (int)(idx / DMODEL), d = (int)(idx % DMODEL);
    int t = tok % SEQ;
    x[idx] = emb[(size_t)ids[tok] * DMODEL + d] + pos[(size_t)t * DMODEL + d];
  }
}

__global__ void k_layernorm(const float* __restrict__ x, const float* __restrict__ g,
                            const float* __restrict__ b, float* __restrict__ out) {
  __shared__ float red[256];
  const int row = blockIdx.x, tid = threadIdx.x;
  const float* xr = x + (size_t)row * DMODEL;
  float s = 0.f;
  for (int i = tid; i < DMODEL; i += 256) s += xr[i];
  red[tid] = s; __syncthreads();
  for (int o = 128; o > 0; o >>= 1) { if (tid < o) red[tid] += red[tid + o]; __syncthreads(); }
  const float mean = red[0] * (1.f / DMODEL);
  __syncthreads();
  float vs = 0.f;
  for (int i = tid; i < DMODEL; i += 256) { float d = xr[i] - mean; vs += d * d; }
  red[tid] = vs; __syncthreads();
  for (int o = 128; o > 0; o >>= 1) { if (tid < o) red[tid] += red[tid + o]; __syncthreads(); }
  const float rstd = rsqrtf(red[0] * (1.f / DMODEL) + 1e-5f);
  float* orow = out + (size_t)row * DMODEL;
  for (int i = tid; i < DMODEL; i += 256) orow[i] = (xr[i] - mean) * rstd * g[i] + b[i];
}

__global__ void k_reset(unsigned* p) { if (threadIdx.x == 0 && blockIdx.x == 0) *p = 0u; }

__global__ void k_absmax(const float* __restrict__ x, size_t n, unsigned* slot) {
  __shared__ float red[256];
  float m = 0.f;
  for (size_t i = (size_t)blockIdx.x * 256 + threadIdx.x; i < n; i += (size_t)gridDim.x * 256)
    m = fmaxf(m, fabsf(x[i]));
  red[threadIdx.x] = m; __syncthreads();
  for (int o = 128; o > 0; o >>= 1) {
    if (threadIdx.x < o) red[threadIdx.x] = fmaxf(red[threadIdx.x], red[threadIdx.x + o]);
    __syncthreads();
  }
  if (threadIdx.x == 0) atomicMax(slot, __float_as_uint(red[0]));
}

__global__ void k_quant_act(const float* __restrict__ x, size_t n,
                            const unsigned* __restrict__ amax, i8* __restrict__ out) {
  const float am = fmaxf(__uint_as_float(*amax), 1e-5f);
  const float s = 127.f / am;
  for (size_t i = (size_t)blockIdx.x * blockDim.x + threadIdx.x; i < n;
       i += (size_t)gridDim.x * blockDim.x) {
    float v = rintf(x[i] * s);
    v = fminf(fmaxf(v, -128.f), 127.f);
    out[i] = (i8)v;
  }
}

__global__ void k_abssum(const float* __restrict__ w, size_t n, float* __restrict__ part) {
  __shared__ float red[256];
  float s = 0.f;
  for (size_t i = (size_t)blockIdx.x * 256 + threadIdx.x; i < n; i += (size_t)gridDim.x * 256)
    s += fabsf(w[i]);
  red[threadIdx.x] = s; __syncthreads();
  for (int o = 128; o > 0; o >>= 1) {
    if (threadIdx.x < o) red[threadIdx.x] += red[threadIdx.x + o];
    __syncthreads();
  }
  if (threadIdx.x == 0) part[blockIdx.x] = red[0];
}

__global__ void k_finalize_sw(const float* __restrict__ part, int nb, float inv_n, float* sw) {
  __shared__ float red[256];
  float s = 0.f;
  for (int i = threadIdx.x; i < nb; i += 256) s += part[i];
  red[threadIdx.x] = s; __syncthreads();
  for (int o = 128; o > 0; o >>= 1) {
    if (threadIdx.x < o) red[threadIdx.x] += red[threadIdx.x + o];
    __syncthreads();
  }
  if (threadIdx.x == 0) *sw = fmaxf(red[0] * inv_n, 1e-5f);
}

__global__ void k_quant_w(const float* __restrict__ w, size_t n,
                          const float* __restrict__ swp, i8* __restrict__ out) {
  const float inv = 1.f / (*swp);
  for (size_t i = (size_t)blockIdx.x * blockDim.x + threadIdx.x; i < n;
       i += (size_t)gridDim.x * blockDim.x) {
    float v = rintf(w[i] * inv);
    v = fminf(fmaxf(v, -1.f), 1.f);
    out[i] = (i8)v;
  }
}

__global__ void k_to_bf16(const float* __restrict__ x, size_t n, u16* __restrict__ out) {
  for (size_t i = (size_t)blockIdx.x * blockDim.x + threadIdx.x; i < n;
       i += (size_t)gridDim.x * blockDim.x) {
    unsigned u = __float_as_uint(x[i]);
    unsigned r = u + 0x7fffu + ((u >> 16) & 1u);
    out[i] = (u16)(r >> 16);
  }
}

template <int MODE>
__global__ __launch_bounds__(256) void k_gemm_i8(
    const i8* __restrict__ A, const i8* __restrict__ W,
    const float* __restrict__ bias, const float* __restrict__ swp,
    const unsigned* __restrict__ amaxp, float extra,
    const float* __restrict__ resid, void* __restrict__ outp,
    int M, int N, int K) {
  __shared__ alignas(16) i8 As[2][128 * 80];
  __shared__ alignas(16) i8 Bs[2][128 * 80];
  const int tid = threadIdx.x;
  const int wave = tid >> 5, lane = tid & 31;
  const int wm = wave & 3, wn = wave >> 2;
  const int kh = lane >> 4, ln = lane & 15;
  const int m0 = blockIdx.y * 128, n0 = blockIdx.x * 128;
  const int row = tid >> 2, kc = (tid & 3) * 16;

  v8i acc[2][4];
#pragma unroll
  for (int s = 0; s < 2; s++)
#pragma unroll
    for (int j = 0; j < 4; j++) acc[s][j] = (v8i)0;

  {
    const i8* Ag = A + (size_t)(m0 + row) * K + kc;
    const i8* Bg = W + (size_t)(n0 + row) * K + kc;
    *(int4*)(&As[0][row * 80 + kc])        = *(const int4*)(Ag);
    *(int4*)(&As[0][(row + 64) * 80 + kc]) = *(const int4*)(Ag + (size_t)64 * K);
    *(int4*)(&Bs[0][row * 80 + kc])        = *(const int4*)(Bg);
    *(int4*)(&Bs[0][(row + 64) * 80 + kc]) = *(const int4*)(Bg + (size_t)64 * K);
  }
  __syncthreads();

  for (int kb = 0; kb < K; kb += 64) {
    const int cur = (kb >> 6) & 1;
    const bool hasNext = (kb + 64) < K;

    int4 ra0{}, ra1{}, rb0{}, rb1{};
    if (hasNext) {
      const i8* Ag = A + (size_t)(m0 + row) * K + kb + 64 + kc;
      const i8* Bg = W + (size_t)(n0 + row) * K + kb + 64 + kc;
      ra0 = *(const int4*)(Ag);
      ra1 = *(const int4*)(Ag + (size_t)64 * K);
      rb0 = *(const int4*)(Bg);
      rb1 = *(const int4*)(Bg + (size_t)64 * K);
      if ((kb + 128) < K) {
        __builtin_prefetch(Ag + 64, 0, 3);
        __builtin_prefetch(Bg + 64, 0, 3);
      }
    }

    FragI af[2], bf[4];
#pragma unroll
    for (int s = 0; s < 2; s++) {
      const i8* ar = &As[cur][(wm * 32 + s * 16 + ln) * 80 + kh * 16];
      af[s].q[0] = *(const int4*)(ar);
      af[s].q[1] = *(const int4*)(ar + 32);
    }
#pragma unroll
    for (int j = 0; j < 4; j++) {
      const i8* br = &Bs[cur][(wn * 64 + j * 16 + ln) * 80 + kh * 16];
      bf[j].q[0] = *(const int4*)(br);
      bf[j].q[1] = *(const int4*)(br + 32);
    }

#pragma unroll
    for (int s = 0; s < 2; s++)
#pragma unroll
      for (int j = 0; j < 4; j++)
        acc[s][j] = __builtin_amdgcn_wmma_i32_16x16x64_iu8(
            true, af[s].v, true, bf[j].v, acc[s][j], false, false);

    if (hasNext) {
      const int nxt = cur ^ 1;
      *(int4*)(&As[nxt][row * 80 + kc])        = ra0;
      *(int4*)(&As[nxt][(row + 64) * 80 + kc]) = ra1;
      *(int4*)(&Bs[nxt][row * 80 + kc])        = rb0;
      *(int4*)(&Bs[nxt][(row + 64) * 80 + kc]) = rb1;
    }
    __syncthreads();
  }

  const float sw = *swp;
  const float am = fmaxf(__uint_as_float(*amaxp), 1e-5f);
  const float scale = sw * am * (1.f / 127.f);

#pragma unroll
  for (int s = 0; s < 2; s++) {
#pragma unroll
    for (int j = 0; j < 4; j++) {
      const int col = n0 + wn * 64 + j * 16 + ln;
      const float bc = bias ? bias[col] : 0.f;
#pragma unroll
      for (int i = 0; i < 8; i++) {
        const int r = m0 + wm * 32 + s * 16 + i + kh * 8;
        const float val = (float)acc[s][j][i] * scale + bc;
        const size_t idx = (size_t)r * N + col;
        if constexpr (MODE == 0) {
          ((_Float16*)outp)[idx] = (_Float16)(val * extra);
        } else if constexpr (MODE == 1) {
          ((float*)outp)[idx] = 0.5f * val * (1.f + erff(val * 0.70710678118654752f));
        } else {
          ((float*)outp)[idx] = resid[idx] + val;
        }
      }
    }
  }
}

#define QS 72
#define KS 72
#define VS 40
#define PS 40
__global__ __launch_bounds__(256) void k_attn(
    const _Float16* __restrict__ q, const _Float16* __restrict__ k,
    const _Float16* __restrict__ v, float* __restrict__ o) {
  __shared__ alignas(16) _Float16 Qs[128 * QS];
  __shared__ alignas(16) _Float16 Ks[32 * KS];
  __shared__ alignas(16) _Float16 Vt[64 * VS];
  __shared__ alignas(16) _Float16 Ps[8 * 16 * PS];
  const int tid = threadIdx.x, wave = tid >> 5, lane = tid & 31;
  const int kh = lane >> 4, ln = lane & 15;
  const int bh = blockIdx.y, b = bh >> 4, hh = bh & 15;
  const int qt0 = blockIdx.x * 128;
  const size_t base = (size_t)b * SEQ * DMODEL + (size_t)hh * HDIM;

#pragma unroll
  for (int it = 0; it < 4; it++) {
    int c = tid + it * 256;
    int r = c >> 3, cc = c & 7;
    *(int4*)(Qs + r * QS + cc * 8) =
        *(const int4*)(q + base + (size_t)(qt0 + r) * DMODEL + cc * 8);
  }
  __syncthreads();

  FragH a0, a1;
  {
    const _Float16* ar = Qs + (wave * 16 + ln) * QS + kh * 8;
    a0.q[0] = *(const int4*)(ar);      a0.q[1] = *(const int4*)(ar + 16);
    a1.q[0] = *(const int4*)(ar + 32); a1.q[1] = *(const int4*)(ar + 48);
  }

  float mrow[8], lrow[8];
  v8f O[4];
#pragma unroll
  for (int i = 0; i < 8; i++) { mrow[i] = -1e30f; lrow[i] = 0.f; }
#pragma unroll
  for (int j = 0; j < 4; j++) O[j] = (v8f)0.f;

  for (int kt = 0; kt < SEQ; kt += 32) {
    __syncthreads();
    {
      int r = tid >> 3, cc = tid & 7;
      *(int4*)(Ks + r * KS + cc * 8) =
          *(const int4*)(k + base + (size_t)(kt + r) * DMODEL + cc * 8);
      union { int4 q4; _Float16 h[8]; } tv;
      tv.q4 = *(const int4*)(v + base + (size_t)(kt + r) * DMODEL + cc * 8);
#pragma unroll
      for (int i = 0; i < 8; i++) Vt[(cc * 8 + i) * VS + r] = tv.h[i];
      if (kt + 32 < SEQ) {
        __builtin_prefetch(k + base + (size_t)(kt + 32 + r) * DMODEL + cc * 8, 0, 3);
        __builtin_prefetch(v + base + (size_t)(kt + 32 + r) * DMODEL + cc * 8, 0, 3);
      }
    }
    __syncthreads();

    v8f S[2];
#pragma unroll
    for (int j = 0; j < 2; j++) {
      FragH b0, b1;
      const _Float16* br = Ks + (j * 16 + ln) * KS;
      b0.q[0] = *(const int4*)(br + kh * 8);      b0.q[1] = *(const int4*)(br + 16 + kh * 8);
      b1.q[0] = *(const int4*)(br + 32 + kh * 8); b1.q[1] = *(const int4*)(br + 48 + kh * 8);
      v8f z = (v8f)0.f;
      z = __builtin_amdgcn_wmma_f32_16x16x32_f16(false, a0.v, false, b0.v, (short)0, z, false, false);
      S[j] = __builtin_amdgcn_wmma_f32_16x16x32_f16(false, a1.v, false, b1.v, (short)0, z, false, false);
    }

    _Float16* pw = Ps + wave * 16 * PS;
#pragma unroll
    for (int i = 0; i < 8; i++) {
      float s0 = S[0][i], s1 = S[1][i];
      float rv = fmaxf(s0, s1);
#pragma unroll
      for (int off = 1; off < 16; off <<= 1) rv = fmaxf(rv, __shfl_xor(rv, off, 32));
      float mn = fmaxf(mrow[i], rv);
      float alpha = __expf(mrow[i] - mn);
      float p0 = __expf(s0 - mn), p1 = __expf(s1 - mn);
      float rs = p0 + p1;
#pragma unroll
      for (int off = 1; off < 16; off <<= 1) rs += __shfl_xor(rs, off, 32);
      lrow[i] = lrow[i] * alpha + rs;
      mrow[i] = mn;
#pragma unroll
      for (int j = 0; j < 4; j++) O[j][i] *= alpha;
      int prow = i + kh * 8;
      pw[prow * PS + ln]      = (_Float16)p0;
      pw[prow * PS + 16 + ln] = (_Float16)p1;
    }

    FragH pa;
    {
      const _Float16* ar = pw + ln * PS + kh * 8;
      pa.q[0] = *(const int4*)(ar); pa.q[1] = *(const int4*)(ar + 16);
    }
#pragma unroll
    for (int j = 0; j < 4; j++) {
      FragH bv;
      const _Float16* br = Vt + (j * 16 + ln) * VS;
      bv.q[0] = *(const int4*)(br + kh * 8); bv.q[1] = *(const int4*)(br + 16 + kh * 8);
      O[j] = __builtin_amdgcn_wmma_f32_16x16x32_f16(false, pa.v, false, bv.v, (short)0, O[j], false, false);
    }
  }

#pragma unroll
  for (int i = 0; i < 8; i++) {
    const float inv = 1.f / lrow[i];
    const int r = qt0 + wave * 16 + i + kh * 8;
#pragma unroll
    for (int j = 0; j < 4; j++)
      o[((size_t)b * SEQ + r) * DMODEL + hh * HDIM + j * 16 + ln] = O[j][i] * inv;
  }
}

__global__ __launch_bounds__(256) void k_gemm_bf16(
    const u16* __restrict__ A, const u16* __restrict__ W,
    float* __restrict__ out, int M, int N, int K) {
  __shared__ alignas(16) u16 As[2][128 * 40];
  __shared__ alignas(16) u16 Bs[2][128 * 40];
  const int tid = threadIdx.x;
  const int wave = tid >> 5, lane = tid & 31;
  const int wm = wave & 3, wn = wave >> 2;
  const int kh = lane >> 4, ln = lane & 15;
  const int m0 = blockIdx.y * 128, n0 = blockIdx.x * 128;
  const int row = tid >> 2, kc = (tid & 3) * 8;

  v8f acc[2][4];
#pragma unroll
  for (int s = 0; s < 2; s++)
#pragma unroll
    for (int j = 0; j < 4; j++) acc[s][j] = (v8f)0.f;

  {
    const u16* Ag = A + (size_t)(m0 + row) * K + kc;
    const u16* Bg = W + (size_t)(n0 + row) * K + kc;
    *(int4*)(&As[0][row * 40 + kc])        = *(const int4*)(Ag);
    *(int4*)(&As[0][(row + 64) * 40 + kc]) = *(const int4*)(Ag + (size_t)64 * K);
    *(int4*)(&Bs[0][row * 40 + kc])        = *(const int4*)(Bg);
    *(int4*)(&Bs[0][(row + 64) * 40 + kc]) = *(const int4*)(Bg + (size_t)64 * K);
  }
  __syncthreads();

  for (int kb = 0; kb < K; kb += 32) {
    const int cur = (kb >> 5) & 1;
    const bool hasNext = (kb + 32) < K;

    int4 ra0{}, ra1{}, rb0{}, rb1{};
    if (hasNext) {
      const u16* Ag = A + (size_t)(m0 + row) * K + kb + 32 + kc;
      const u16* Bg = W + (size_t)(n0 + row) * K + kb + 32 + kc;
      ra0 = *(const int4*)(Ag);
      ra1 = *(const int4*)(Ag + (size_t)64 * K);
      rb0 = *(const int4*)(Bg);
      rb1 = *(const int4*)(Bg + (size_t)64 * K);
      if ((kb + 64) < K) {
        __builtin_prefetch(Ag + 32, 0, 3);
        __builtin_prefetch(Bg + 32, 0, 3);
      }
    }

    FragB af[2], bf[4];
#pragma unroll
    for (int s = 0; s < 2; s++) {
      const u16* ar = &As[cur][(wm * 32 + s * 16 + ln) * 40 + kh * 8];
      af[s].q[0] = *(const int4*)(ar); af[s].q[1] = *(const int4*)(ar + 16);
    }
#pragma unroll
    for (int j = 0; j < 4; j++) {
      const u16* br = &Bs[cur][(wn * 64 + j * 16 + ln) * 40 + kh * 8];
      bf[j].q[0] = *(const int4*)(br); bf[j].q[1] = *(const int4*)(br + 16);
    }

#pragma unroll
    for (int s = 0; s < 2; s++)
#pragma unroll
      for (int j = 0; j < 4; j++)
        acc[s][j] = __builtin_amdgcn_wmma_f32_16x16x32_bf16(
            false, af[s].v, false, bf[j].v, (short)0, acc[s][j], false, false);

    if (hasNext) {
      const int nxt = cur ^ 1;
      *(int4*)(&As[nxt][row * 40 + kc])        = ra0;
      *(int4*)(&As[nxt][(row + 64) * 40 + kc]) = ra1;
      *(int4*)(&Bs[nxt][row * 40 + kc])        = rb0;
      *(int4*)(&Bs[nxt][(row + 64) * 40 + kc]) = rb1;
    }
    __syncthreads();
  }

#pragma unroll
  for (int s = 0; s < 2; s++)
#pragma unroll
    for (int j = 0; j < 4; j++) {
      const int col = n0 + wn * 64 + j * 16 + ln;
#pragma unroll
      for (int i = 0; i < 8; i++) {
        const int r = m0 + wm * 32 + s * 16 + i + kh * 8;
        out[(size_t)r * N + col] = acc[s][j][i];
      }
    }
}

extern "C" void kernel_launch(void* const* d_in, const int* in_sizes, int n_in,
                              void* d_out, int out_size, void* d_ws, size_t ws_size,
                              hipStream_t stream) {
  (void)in_sizes; (void)n_in; (void)out_size; (void)ws_size;
  const int*   ids   = (const int*)d_in[0];
  const float* emb   = (const float*)d_in[1];
  const float* pos   = (const float*)d_in[2];
  const float* ag    = (const float*)d_in[3];
  const float* abp   = (const float*)d_in[4];
  const float* wq    = (const float*)d_in[5];
  const float* bq    = (const float*)d_in[6];
  const float* wk    = (const float*)d_in[7];
  const float* bk    = (const float*)d_in[8];
  const float* wv    = (const float*)d_in[9];
  const float* bvp   = (const float*)d_in[10];
  const float* wo    = (const float*)d_in[11];
  const float* bo    = (const float*)d_in[12];
  const float* fg    = (const float*)d_in[13];
  const float* fb    = (const float*)d_in[14];
  const float* w1    = (const float*)d_in[15];
  const float* b1    = (const float*)d_in[16];
  const float* w2    = (const float*)d_in[17];
  const float* b2    = (const float*)d_in[18];
  const float* lnfg  = (const float*)d_in[19];
  const float* lnfb  = (const float*)d_in[20];
  const float* headw = (const float*)d_in[21];

  char* ws = (char*)d_ws;
  size_t off = 0;
  auto alloc = [&](size_t bytes) -> char* {
    char* p = ws + off; off += (bytes + 255) & ~(size_t)255; return p;
  };
  float*    x    = (float*)alloc((size_t)NTOK * DMODEL * 4);
  float*    xl   = (float*)alloc((size_t)NTOK * DMODEL * 4);
  i8*       a8   = (i8*)   alloc((size_t)NTOK * FDIM);
  _Float16* qf   = (_Float16*)alloc((size_t)NTOK * DMODEL * 2);
  _Float16* kf   = (_Float16*)alloc((size_t)NTOK * DMODEL * 2);
  _Float16* vf   = (_Float16*)alloc((size_t)NTOK * DMODEL * 2);
  float*    ao   = (float*)alloc((size_t)NTOK * DMODEL * 4);
  float*    hbuf = (float*)alloc((size_t)NTOK * FDIM * 4);
  i8*       w8   = (i8*)   alloc((size_t)FDIM * DMODEL);
  u16*      xbf  = (u16*)  alloc((size_t)NTOK * DMODEL * 2);
  u16*      hwbf = (u16*)  alloc((size_t)VOCAB * DMODEL * 2);
  unsigned* amax = (unsigned*)alloc(256);
  float*    sw   = (float*)alloc(256);
  float*    part = (float*)alloc(1024 * 4);

  const dim3 blk(256);
  const dim3 gD(DMODEL / 128, NTOK / 128);
  const dim3 gF(FDIM / 128, NTOK / 128);

  k_to_bf16<<<2048, blk, 0, stream>>>(headw, (size_t)VOCAB * DMODEL, hwbf);
  k_embed<<<1024, blk, 0, stream>>>(ids, emb, pos, x);

  auto do_absmax = [&](const float* p, size_t n) {
    k_reset<<<1, 1, 0, stream>>>(amax);
    k_absmax<<<512, blk, 0, stream>>>(p, n, amax);
  };
  auto do_wquant = [&](const float* w, size_t n) {
    k_abssum<<<1024, blk, 0, stream>>>(w, n, part);
    k_finalize_sw<<<1, blk, 0, stream>>>(part, 1024, 1.f / (float)n, sw);
    k_quant_w<<<1024, blk, 0, stream>>>(w, n, sw, w8);
  };

  for (int l = 0; l < NLAYER; l++) {
    const size_t lD = (size_t)l * DMODEL, lF = (size_t)l * FDIM;
    const size_t lDD = (size_t)l * DMODEL * DMODEL, lFD = (size_t)l * FDIM * DMODEL;

    k_layernorm<<<NTOK, blk, 0, stream>>>(x, ag + lD, abp + lD, xl);
    do_absmax(xl, (size_t)NTOK * DMODEL);
    k_quant_act<<<1024, blk, 0, stream>>>(xl, (size_t)NTOK * DMODEL, amax, a8);

    do_wquant(wq + lDD, (size_t)DMODEL * DMODEL);
    k_gemm_i8<0><<<gD, blk, 0, stream>>>(a8, w8, bq + lD, sw, amax, 0.125f, nullptr, qf,
                                         NTOK, DMODEL, DMODEL);
    do_wquant(wk + lDD, (size_t)DMODEL * DMODEL);
    k_gemm_i8<0><<<gD, blk, 0, stream>>>(a8, w8, bk + lD, sw, amax, 1.0f, nullptr, kf,
                                         NTOK, DMODEL, DMODEL);
    do_wquant(wv + lDD, (size_t)DMODEL * DMODEL);
    k_gemm_i8<0><<<gD, blk, 0, stream>>>(a8, w8, bvp + lD, sw, amax, 1.0f, nullptr, vf,
                                         NTOK, DMODEL, DMODEL);

    k_attn<<<dim3(SEQ / 128, BATCH * NHEAD), blk, 0, stream>>>(qf, kf, vf, ao);

    do_absmax(ao, (size_t)NTOK * DMODEL);
    k_quant_act<<<1024, blk, 0, stream>>>(ao, (size_t)NTOK * DMODEL, amax, a8);
    do_wquant(wo + lDD, (size_t)DMODEL * DMODEL);
    k_gemm_i8<2><<<gD, blk, 0, stream>>>(a8, w8, bo + lD, sw, amax, 1.0f, x, x,
                                         NTOK, DMODEL, DMODEL);

    k_layernorm<<<NTOK, blk, 0, stream>>>(x, fg + lD, fb + lD, xl);
    do_absmax(xl, (size_t)NTOK * DMODEL);
    k_quant_act<<<1024, blk, 0, stream>>>(xl, (size_t)NTOK * DMODEL, amax, a8);
    do_wquant(w1 + lFD, (size_t)FDIM * DMODEL);
    k_gemm_i8<1><<<gF, blk, 0, stream>>>(a8, w8, b1 + lF, sw, amax, 1.0f, nullptr, hbuf,
                                         NTOK, FDIM, DMODEL);
    do_absmax(hbuf, (size_t)NTOK * FDIM);
    k_quant_act<<<2048, blk, 0, stream>>>(hbuf, (size_t)NTOK * FDIM, amax, a8);
    do_wquant(w2 + lFD, (size_t)DMODEL * FDIM);
    k_gemm_i8<2><<<gD, blk, 0, stream>>>(a8, w8, b2 + lD, sw, amax, 1.0f, x, x,
                                         NTOK, DMODEL, FDIM);
  }

  k_layernorm<<<NTOK, blk, 0, stream>>>(x, lnfg, lnfb, xl);
  k_to_bf16<<<1024, blk, 0, stream>>>(xl, (size_t)NTOK * DMODEL, xbf);
  k_gemm_bf16<<<dim3(VOCAB / 128, NTOK / 128), blk, 0, stream>>>(
      xbf, hwbf, (float*)d_out, NTOK, VOCAB, DMODEL);
}
